// LatentMultiHeadAttention_15504831939080
// MI455X (gfx1250) — matrix operand order repaired and checked by reading, not compiled, not run on hardware
//
#include <hip/hip_runtime.h>
#include <hip/hip_bf16.h>
#include <stdint.h>

typedef __bf16 bf16_t;
typedef __attribute__((ext_vector_type(8)))  __bf16 v8bf;
typedef __attribute__((ext_vector_type(16))) __bf16 v16bf;
typedef __attribute__((ext_vector_type(8)))  float  v8f;

union FragBF {
  v16bf v;
  v8bf  h[2];
};

__device__ __forceinline__ bf16_t f2bf(float f) {
  uint32_t u = __builtin_bit_cast(uint32_t, f);
  uint32_t r = (u + 0x7FFFu + ((u >> 16) & 1u)) >> 16;
  uint16_t s = (uint16_t)r;
  return __builtin_bit_cast(bf16_t, s);
}

// ---------------------------------------------------------------------------
// fp32 -> bf16 elementwise convert
// ---------------------------------------------------------------------------
__global__ __launch_bounds__(256)
void cvt_f32_bf16_kernel(const float* __restrict__ src, bf16_t* __restrict__ dst, int n) {
  int i = blockIdx.x * 256 + threadIdx.x;
  if (i < n) dst[i] = f2bf(src[i]);
}

// ---------------------------------------------------------------------------
// Batched tiled transpose + convert: src fp32 [batch][R][C] -> dst bf16 [batch][C][R]
// ---------------------------------------------------------------------------
__global__ __launch_bounds__(256)
void transpose_cvt_kernel(const float* __restrict__ src, bf16_t* __restrict__ dst,
                          int R, int C) {
  __shared__ float tile[32][33];
  const int batch = blockIdx.z;
  src += (size_t)batch * R * C;
  dst += (size_t)batch * R * C;
  const int c0 = blockIdx.x * 32;
  const int r0 = blockIdx.y * 32;
  const int tx = threadIdx.x;   // 0..31
  const int ty = threadIdx.y;   // 0..7
#pragma unroll
  for (int i = 0; i < 32; i += 8)
    tile[ty + i][tx] = src[(size_t)(r0 + ty + i) * C + (c0 + tx)];
  __syncthreads();
#pragma unroll
  for (int i = 0; i < 32; i += 8)
    dst[(size_t)(c0 + ty + i) * R + (r0 + tx)] = f2bf(tile[tx][ty + i]);
}

// ---------------------------------------------------------------------------
// bf16 WMMA GEMM: C[M,N] = A[M,K] * Bt[N,K]^T + bias[N]
//   OUT_MODE 0: bf16 row-major [M][N]
//   OUT_MODE 1: f32  row-major [M][N]
//   OUT_MODE 2: bf16 per-head transposed store Vt[b][h][d][t]
//               (row = b*2048+t, col = h*128+d; hardcoded B=2,T=2048,H=16,Dh=128)
// Block: 256 threads = 8 waves; tile 128x128, K-step 32; each wave: 32x64.
// Data movement: double-buffered GLOBAL_LOAD_ASYNC_TO_LDS_B128 (ASYNCcnt).
// ---------------------------------------------------------------------------
template<int OUT_MODE>
__global__ __launch_bounds__(256)
void gemm_bf16_kernel(const bf16_t* __restrict__ A,
                      const bf16_t* __restrict__ Bt,
                      const float* __restrict__ bias,
                      void* __restrict__ Cout,
                      int M, int N, int K)
{
  __shared__ bf16_t sA[2][128][40];   // padded rows: conflict-free b128 reads
  __shared__ bf16_t sB[2][128][40];

  const int tid  = threadIdx.x;
  const int lane = tid & 31;
  const int wave = tid >> 5;
  const int ln   = lane & 15;
  const int hi   = lane >> 4;
  const int wm   = wave >> 1;     // 0..3  (M direction)
  const int wn   = wave & 1;      // 0..1  (N direction)

  const int row0 = blockIdx.y * 128;
  const int col0 = blockIdx.x * 128;

  const int lrow = tid >> 1;          // 0..127
  const int lcol = (tid & 1) << 4;    // 0 or 16

  v8f acc[2][4] = {};

  const bf16_t* gA = A  + (size_t)(row0 + lrow) * K + lcol;
  const bf16_t* gB = Bt + (size_t)(col0 + lrow) * K + lcol;

  // Per-thread async copy of its 32B slice of the A and B tiles.
  // INST_OFFSET is added to both the global and LDS addresses (ISA 10.x),
  // so one base covers both 16B chunks via offset:16.
  auto copy_async = [&](int buf, int k0) {
    uint32_t la = (uint32_t)(size_t)&sA[buf][lrow][lcol];
    uint32_t lb = (uint32_t)(size_t)&sB[buf][lrow][lcol];
    uint64_t ga = (uint64_t)(size_t)(gA + k0);
    uint64_t gb = (uint64_t)(size_t)(gB + k0);
    asm volatile(
        "global_load_async_to_lds_b128 %0, %2, off\n\t"
        "global_load_async_to_lds_b128 %0, %2, off offset:16\n\t"
        "global_load_async_to_lds_b128 %1, %3, off\n\t"
        "global_load_async_to_lds_b128 %1, %3, off offset:16"
        :: "v"(la), "v"(lb), "v"(ga), "v"(gb)
        : "memory");
  };

  copy_async(0, 0);   // prologue: tile 0 in flight
  int buf = 0;

  for (int k0 = 0; k0 < K; k0 += 32) {
    __syncthreads();  // (a) all waves done reading buf^1 before we overwrite it
    const bool has_next = (k0 + 32) < K;
    if (has_next) {
      copy_async(buf ^ 1, k0 + 32);
      // <=4 outstanding: the 4 copies for the *current* buffer have landed
      asm volatile("s_wait_asynccnt 4" ::: "memory");
    } else {
      asm volatile("s_wait_asynccnt 0" ::: "memory");
    }
    __syncthreads();  // (b) every wave waited on its own copies -> tile visible

    FragBF fa[2];
#pragma unroll
    for (int mt = 0; mt < 2; ++mt) {
      const int r = wm * 32 + mt * 16 + ln;
      fa[mt].h[0] = *(const v8bf*)&sA[buf][r][hi * 8];
      fa[mt].h[1] = *(const v8bf*)&sA[buf][r][16 + hi * 8];
    }
#pragma unroll
    for (int nt = 0; nt < 4; ++nt) {
      const int c = wn * 64 + nt * 16 + ln;
      FragBF fb;
      fb.h[0] = *(const v8bf*)&sB[buf][c][hi * 8];
      fb.h[1] = *(const v8bf*)&sB[buf][c][16 + hi * 8];
#pragma unroll
      for (int mt = 0; mt < 2; ++mt) {
        acc[mt][nt] = __builtin_amdgcn_wmma_f32_16x16x32_bf16(
            false, fa[mt].v, false, fb.v, (short)0, acc[mt][nt], false, false);
      }
    }
    buf ^= 1;
  }

#pragma unroll
  for (int mt = 0; mt < 2; ++mt) {
#pragma unroll
    for (int nt = 0; nt < 4; ++nt) {
      const int col = col0 + wn * 64 + nt * 16 + ln;
      const float bvv = bias[col];
      const int rbase = row0 + wm * 32 + mt * 16 + hi * 8;
#pragma unroll
      for (int r = 0; r < 8; ++r) {
        const float o = acc[mt][nt][r] + bvv;
        const int row = rbase + r;
        if (OUT_MODE == 1) {
          ((float*)Cout)[(size_t)row * N + col] = o;
        } else if (OUT_MODE == 0) {
          ((bf16_t*)Cout)[(size_t)row * N + col] = f2bf(o);
        } else {
          const int bb = row >> 11, t = row & 2047;
          const int hh = col >> 7,  d = col & 127;
          ((bf16_t*)Cout)[(((size_t)bb * 16 + hh) * 128 + d) * 2048 + t] = f2bf(o);
        }
      }
    }
  }
}

// ---------------------------------------------------------------------------
// Flash attention (non-causal, full softmax).
//   Q, K: bf16 [B][T][H*128] (row-major per token)
//   Vt:   bf16 [B][H][128][T] (per-head transposed -> contiguous keys per d)
//   Out:  bf16 [B][T][H*128]
// Grid: (T/128, H, B). Block 256 = 8 waves; each wave owns 16 query rows.
// ---------------------------------------------------------------------------
__global__ __launch_bounds__(256)
void flash_attn_kernel(const bf16_t* __restrict__ Qm,
                       const bf16_t* __restrict__ Km,
                       const bf16_t* __restrict__ Vt,
                       bf16_t* __restrict__ Om)
{
  __shared__ bf16_t sp[8][16][72];   // per-wave 16x64 P tile, padded rows

  const int lane = threadIdx.x & 31;
  const int wave = threadIdx.x >> 5;
  const int ln = lane & 15;
  const int hi = lane >> 4;

  const int h  = blockIdx.y;
  const int b  = blockIdx.z;
  const int q0 = blockIdx.x * 128 + wave * 16;

  // Q fragments: 16 rows x 128 (K split in 4 chunks of 32), kept resident.
  FragBF fq[4];
  {
    const bf16_t* qrow = Qm + (size_t)(b * 2048 + q0 + ln) * 2048 + h * 128;
#pragma unroll
    for (int kk = 0; kk < 4; ++kk) {
      fq[kk].h[0] = *(const v8bf*)(qrow + kk * 32 + hi * 8);
      fq[kk].h[1] = *(const v8bf*)(qrow + kk * 32 + 16 + hi * 8);
    }
  }

  v8f O[8] = {};
  float Mr[8], Lr[8];
#pragma unroll
  for (int r = 0; r < 8; ++r) { Mr[r] = -1e30f; Lr[r] = 0.f; }

  const bf16_t* kbase = Km + (size_t)b * 2048 * 2048 + h * 128;
  const bf16_t* vbase = Vt + (size_t)(b * 16 + h) * 128 * 2048;
  bf16_t (*myp)[72] = sp[wave];

  const float scale = 0.08838834764831845f;  // 1/sqrt(128)

  for (int k0 = 0; k0 < 2048; k0 += 64) {
    // ---- S = Q K^T (16 x 64 tile) ----
    v8f s[4] = {};
#pragma unroll
    for (int nt = 0; nt < 4; ++nt) {
      const bf16_t* krow = kbase + (size_t)(k0 + nt * 16 + ln) * 2048;
#pragma unroll
      for (int kk = 0; kk < 4; ++kk) {
        FragBF fk;
        fk.h[0] = *(const v8bf*)(krow + kk * 32 + hi * 8);
        fk.h[1] = *(const v8bf*)(krow + kk * 32 + 16 + hi * 8);
        s[nt] = __builtin_amdgcn_wmma_f32_16x16x32_bf16(
            false, fq[kk].v, false, fk.v, (short)0, s[nt], false, false);
      }
    }

    // ---- online softmax (row = hi*8 + r, striped over 16 lanes) ----
    float cf[8];
#pragma unroll
    for (int r = 0; r < 8; ++r) {
#pragma unroll
      for (int nt = 0; nt < 4; ++nt) s[nt][r] *= scale;
      float mx = fmaxf(fmaxf(s[0][r], s[1][r]), fmaxf(s[2][r], s[3][r]));
#pragma unroll
      for (int d = 1; d < 16; d <<= 1) mx = fmaxf(mx, __shfl_xor(mx, d, 16));
      const float newM = fmaxf(Mr[r], mx);
      cf[r] = __expf(Mr[r] - newM);
      Mr[r] = newM;
      float rs = 0.f;
#pragma unroll
      for (int nt = 0; nt < 4; ++nt) {
        const float p = __expf(s[nt][r] - newM);
        s[nt][r] = p;
        rs += p;
      }
#pragma unroll
      for (int d = 1; d < 16; d <<= 1) rs += __shfl_xor(rs, d, 16);
      Lr[r] = Lr[r] * cf[r] + rs;
    }
#pragma unroll
    for (int dt = 0; dt < 8; ++dt)
#pragma unroll
      for (int r = 0; r < 8; ++r) O[dt][r] *= cf[r];

    // ---- D-layout -> A-layout via per-wave LDS tile ----
#pragma unroll
    for (int nt = 0; nt < 4; ++nt)
#pragma unroll
      for (int r = 0; r < 8; ++r)
        myp[hi * 8 + r][nt * 16 + ln] = f2bf(s[nt][r]);
    asm volatile("s_wait_dscnt 0" ::: "memory");

    // ---- O += P V ----
#pragma unroll
    for (int kk = 0; kk < 2; ++kk) {
      FragBF fp;
      fp.h[0] = *(const v8bf*)&myp[ln][kk * 32 + hi * 8];
      fp.h[1] = *(const v8bf*)&myp[ln][kk * 32 + 16 + hi * 8];
#pragma unroll
      for (int dt = 0; dt < 8; ++dt) {
        const bf16_t* vrow = vbase + (size_t)(dt * 16 + ln) * 2048 + k0 + kk * 32;
        FragBF fv;
        fv.h[0] = *(const v8bf*)(vrow + hi * 8);
        fv.h[1] = *(const v8bf*)(vrow + 16 + hi * 8);
        O[dt] = __builtin_amdgcn_wmma_f32_16x16x32_bf16(
            false, fp.v, false, fv.v, (short)0, O[dt], false, false);
      }
    }
  }

  // ---- normalize and store ----
#pragma unroll
  for (int r = 0; r < 8; ++r) {
    const float inv = 1.0f / Lr[r];
    bf16_t* orow = Om + (size_t)(b * 2048 + q0 + hi * 8 + r) * 2048 + h * 128;
#pragma unroll
    for (int dt = 0; dt < 8; ++dt)
      orow[dt * 16 + ln] = f2bf(O[dt][r] * inv);
  }
}

// ---------------------------------------------------------------------------
extern "C" void kernel_launch(void* const* d_in, const int* in_sizes, int n_in,
                              void* d_out, int out_size, void* d_ws, size_t ws_size,
                              hipStream_t stream)
{
  (void)in_sizes; (void)n_in; (void)out_size; (void)ws_size;

  const float* x  = (const float*)d_in[0];
  const float* Wq = (const float*)d_in[1];
  const float* bq = (const float*)d_in[2];
  const float* Wl = (const float*)d_in[3];
  const float* bl = (const float*)d_in[4];
  const float* Wk = (const float*)d_in[5];
  const float* bk = (const float*)d_in[6];
  const float* Wv = (const float*)d_in[7];
  const float* bv = (const float*)d_in[8];
  const float* Wo = (const float*)d_in[9];
  const float* bo = (const float*)d_in[10];

  char* ws = (char*)d_ws;
  size_t off = 0;
  auto carve = [&](size_t bytes) -> char* {
    char* p = ws + off;
    off += (bytes + 255) & ~(size_t)255;
    return p;
  };
  bf16_t* xb   = (bf16_t*)carve((size_t)4096 * 2048 * 2);  // x bf16
  bf16_t* WqT  = (bf16_t*)carve((size_t)2048 * 2048 * 2);  // Wq^T
  bf16_t* WlT  = (bf16_t*)carve((size_t)512  * 2048 * 2);  // Wl^T
  bf16_t* WkT  = (bf16_t*)carve((size_t)16 * 128 * 512 * 2); // Wk [H][Dh][L]
  bf16_t* WvT  = (bf16_t*)carve((size_t)16 * 128 * 512 * 2); // Wv [H][Dh][L]
  bf16_t* WoT  = (bf16_t*)carve((size_t)2048 * 2048 * 2);  // Wo^T
  bf16_t* Qb   = (bf16_t*)carve((size_t)4096 * 2048 * 2);  // Q   [B][T][H*128]
  bf16_t* Latb = (bf16_t*)carve((size_t)4096 * 512  * 2);  // latent
  bf16_t* Kb   = (bf16_t*)carve((size_t)4096 * 2048 * 2);  // K   [B][T][H*128]
  bf16_t* Vtb  = (bf16_t*)carve((size_t)4096 * 2048 * 2);  // V^T [B][H][128][T]
  bf16_t* Ab   = (bf16_t*)carve((size_t)4096 * 2048 * 2);  // attn out bf16

  // --- precision conversion / weight transposes ---
  {
    const int n = 4096 * 2048;
    cvt_f32_bf16_kernel<<<(n + 255) / 256, 256, 0, stream>>>(x, xb, n);
  }
  transpose_cvt_kernel<<<dim3(64, 64, 1),  dim3(32, 8), 0, stream>>>(Wq, WqT, 2048, 2048);
  transpose_cvt_kernel<<<dim3(16, 64, 1),  dim3(32, 8), 0, stream>>>(Wl, WlT, 2048, 512);
  transpose_cvt_kernel<<<dim3(4, 16, 16),  dim3(32, 8), 0, stream>>>(Wk, WkT, 512, 128);
  transpose_cvt_kernel<<<dim3(4, 16, 16),  dim3(32, 8), 0, stream>>>(Wv, WvT, 512, 128);
  transpose_cvt_kernel<<<dim3(64, 64, 1),  dim3(32, 8), 0, stream>>>(Wo, WoT, 2048, 2048);

  // --- projections (bf16 WMMA GEMMs, async-LDS double buffered) ---
  gemm_bf16_kernel<0><<<dim3(16, 32), 256, 0, stream>>>(xb,   WqT, bq, Qb,   4096, 2048, 2048);
  gemm_bf16_kernel<0><<<dim3(4,  32), 256, 0, stream>>>(xb,   WlT, bl, Latb, 4096, 512,  2048);
  gemm_bf16_kernel<0><<<dim3(16, 32), 256, 0, stream>>>(Latb, WkT, bk, Kb,   4096, 2048, 512);
  gemm_bf16_kernel<2><<<dim3(16, 32), 256, 0, stream>>>(Latb, WvT, bv, Vtb,  4096, 2048, 512);

  // --- attention ---
  flash_attn_kernel<<<dim3(16, 16, 2), 256, 0, stream>>>(Qb, Kb, Vtb, Ab);

  // --- output projection (f32 result) ---
  gemm_bf16_kernel<1><<<dim3(16, 32), 256, 0, stream>>>(Ab, WoT, bo, d_out, 4096, 2048, 2048);
}
